// MLACpu_79551384256988
// MI455X (gfx1250) — matrix operand order repaired and checked by reading, not compiled, not run on hardware
//
#include <hip/hip_runtime.h>

typedef __bf16 bf16;
typedef __attribute__((ext_vector_type(16))) __bf16 v16bf;
typedef __attribute__((ext_vector_type(8)))  __bf16 v8bf;
typedef __attribute__((ext_vector_type(8)))  float  v8f;

#define DEVI static __device__ __forceinline__

namespace {
constexpr int S_LEN = 2048;
constexpr int HEADS = 16;
constexpr int DN    = 128;
constexpr int DR    = 64;
constexpr int DV    = 128;
constexpr int QL    = 1536;
constexpr int CL    = 512;
constexpr int QDIM  = DN + DR;        // 192
constexpr int QROWS = HEADS * QDIM;   // 3072
constexpr float SM_SCALE = 0.07216878364870323f; // 1/sqrt(192)
constexpr float MASK_NEG = -10000.0f;
}

union Frag { v16bf v; v8bf h[2]; };

DEVI v8f vzero() {
  v8f z;
#pragma unroll
  for (int i = 0; i < 8; ++i) z[i] = 0.0f;
  return z;
}

// A fragment (16x32 bf16, MxK) from row-major (M,K) storage.
// lane<16: M=lane, elems0-7 = K[k0..k0+7],  elems8-15 = K[k0+16..k0+23]
// lane>=16: M=lane-16, elems0-7 = K[k0+8..], elems8-15 = K[k0+24..]
DEVI v16bf load_a_frag(const bf16* __restrict__ base, int ld, int m0, int k0, int lane) {
  int hi = lane >> 4, lo = lane & 15;
  const bf16* p = base + (size_t)(m0 + lo) * ld + k0 + 8 * hi;
  Frag f;
  f.h[0] = *reinterpret_cast<const v8bf*>(p);
  f.h[1] = *reinterpret_cast<const v8bf*>(p + 16);
  return f.v;
}

// B fragment (32x16 bf16, KxN) from N-major storage (row n holds K contiguous).
// lane<16: N=lane, elems0-7 = K[k0..k0+7], elems8-15 = K[k0+16..k0+23]; lane>=16: N=lane-16, K[k0+8..], K[k0+24..]
DEVI v16bf load_b_frag(const bf16* __restrict__ base, int ld, int n0, int k0, int lane) {
  int hi = lane >> 4, lo = lane & 15;
  const bf16* p = base + (size_t)(n0 + lo) * ld + k0 + 8 * hi;
  Frag f;
  f.h[0] = *reinterpret_cast<const v8bf*>(p);
  f.h[1] = *reinterpret_cast<const v8bf*>(p + 16);
  return f.v;
}

DEVI v8f wmma_bf16(v16bf a, v16bf b, v8f c) {
  return __builtin_amdgcn_wmma_f32_16x16x32_bf16(false, a, false, b, (short)0, c, false, false);
}

DEVI float rowmax16(float v) {
  v = fmaxf(v, __shfl_xor(v, 1, 16));
  v = fmaxf(v, __shfl_xor(v, 2, 16));
  v = fmaxf(v, __shfl_xor(v, 4, 16));
  v = fmaxf(v, __shfl_xor(v, 8, 16));
  return v;
}
DEVI float rowsum16(float v) {
  v += __shfl_xor(v, 1, 16);
  v += __shfl_xor(v, 2, 16);
  v += __shfl_xor(v, 4, 16);
  v += __shfl_xor(v, 8, 16);
  return v;
}

// 16x64 wave-tile GEMM over K, software-pipelined one K-step ahead so the
// next iteration's 5 global_load_b128 groups are in flight during the 4 WMMAs.
template <int K>
DEVI void gemm16x64(const bf16* __restrict__ A, int lda, int s0,
                    const bf16* __restrict__ B, int ldb, int n0,
                    int lane, v8f acc[4]) {
  v16bf ac = load_a_frag(A, lda, s0, 0, lane);
  v16bf bc[4];
#pragma unroll
  for (int i = 0; i < 4; ++i) bc[i] = load_b_frag(B, ldb, n0 + 16 * i, 0, lane);
#pragma unroll 4
  for (int k0 = 32; k0 < K; k0 += 32) {
    v16bf an = load_a_frag(A, lda, s0, k0, lane);
    v16bf bn[4];
#pragma unroll
    for (int i = 0; i < 4; ++i) bn[i] = load_b_frag(B, ldb, n0 + 16 * i, k0, lane);
#pragma unroll
    for (int i = 0; i < 4; ++i) acc[i] = wmma_bf16(ac, bc[i], acc[i]);
    ac = an;
#pragma unroll
    for (int i = 0; i < 4; ++i) bc[i] = bn[i];
  }
#pragma unroll
  for (int i = 0; i < 4; ++i) acc[i] = wmma_bf16(ac, bc[i], acc[i]);
}

// ---------------- phase 1: dequant + convert ----------------

__global__ void k_dequant_wuq(const float* __restrict__ Wq,
                              const float* __restrict__ sc,
                              bf16* __restrict__ out) {
  int idx = blockIdx.x * blockDim.x + threadIdx.x;   // 3072*1536
  int r = idx / QL, c = idx % QL;
  float v = Wq[idx] * sc[(r >> 7) * (QL >> 7) + (c >> 7)];
  out[idx] = (bf16)v;
}

// WUKV (4096x512): split into WnT[h][c][d] (transposed nope) and Wv[h][d][c]
__global__ void k_dequant_wukv(const float* __restrict__ Wkv,
                               const float* __restrict__ sc,
                               bf16* __restrict__ WnT,
                               bf16* __restrict__ Wv) {
  int idx = blockIdx.x * blockDim.x + threadIdx.x;   // 4096*512
  int r = idx / CL, c = idx % CL;
  float v = Wkv[idx] * sc[(r >> 7) * (CL >> 7) + (c >> 7)];
  int h = r >> 8, j = r & 255;
  if (j < DN) WnT[((size_t)h * CL + c) * DN + j] = (bf16)v;
  else        Wv[((size_t)h * DV + (j - DN)) * CL + c] = (bf16)v;
}

__global__ void k_conv_q(const float* __restrict__ Q, bf16* __restrict__ Qb) {
  int i = blockIdx.x * blockDim.x + threadIdx.x;     // 2048*1536
  Qb[i] = (bf16)Q[i];
}

__global__ void k_conv_kv(const float* __restrict__ KV,
                          bf16* __restrict__ kvb, bf16* __restrict__ kvT) {
  int i = blockIdx.x * blockDim.x + threadIdx.x;     // 2048*512
  int t = i / CL, c = i % CL;
  float v = KV[i];
  kvb[i] = (bf16)v;
  kvT[(size_t)c * S_LEN + t] = (bf16)v;
}

__global__ void k_rope_pe(const float* __restrict__ PE,
                          const float* __restrict__ th,
                          bf16* __restrict__ peb) {
  int i = blockIdx.x * blockDim.x + threadIdx.x;     // 2048*32
  int t = i >> 5, k = i & 31;
  float x0 = PE[t * DR + 2 * k], x1 = PE[t * DR + 2 * k + 1];
  float a = th[t * DR + k], b = th[t * DR + 32 + k];
  peb[t * DR + k]      = (bf16)(x0 * __cosf(a) - x1 * __sinf(a));
  peb[t * DR + 32 + k] = (bf16)(x1 * __cosf(b) + x0 * __sinf(b));
}

// ---------------- phase 2: q = Q @ WUQd^T ----------------

__global__ void __launch_bounds__(128) k_gemm_q(const bf16* __restrict__ Qb,
                                                const bf16* __restrict__ Wq,
                                                float* __restrict__ qfull) {
  int wid  = (blockIdx.x * blockDim.x + threadIdx.x) >> 5;  // 128*48 waves
  int lane = threadIdx.x & 31;
  int mb = wid / 48, nb = wid % 48;
  int s0 = mb * 16, n0 = nb * 64;
  v8f acc[4];
#pragma unroll
  for (int i = 0; i < 4; ++i) acc[i] = vzero();
  gemm16x64<QL>(Qb, QL, s0, Wq, QL, n0, lane, acc);
  int lo = lane & 15, hi = lane >> 4;
#pragma unroll
  for (int i = 0; i < 4; ++i)
#pragma unroll
    for (int r = 0; r < 8; ++r)
      qfull[(size_t)(s0 + r + 8 * hi) * QROWS + n0 + 16 * i + lo] = acc[i][r];
}

// split q into per-head nope (bf16) and rope'd pe (bf16)
__global__ void k_rope_split(const float* __restrict__ qfull,
                             const float* __restrict__ th,
                             bf16* __restrict__ qnope,
                             bf16* __restrict__ qpe) {
  int idx = blockIdx.x * blockDim.x + threadIdx.x;   // 16*2048*32
  int k = idx & 31; int row = idx >> 5;
  int h = row >> 11; int s = row & (S_LEN - 1);
  const float* qr = qfull + (size_t)s * QROWS + h * QDIM;
  bf16* qn = qnope + ((size_t)h * S_LEN + s) * DN;
#pragma unroll
  for (int u = 0; u < 4; ++u) qn[k + 32 * u] = (bf16)qr[k + 32 * u];
  float x0 = qr[DN + 2 * k], x1 = qr[DN + 2 * k + 1];
  float a = th[s * DR + k], b = th[s * DR + 32 + k];
  bf16* qp = qpe + ((size_t)h * S_LEN + s) * DR;
  qp[k]      = (bf16)(x0 * __cosf(a) - x1 * __sinf(a));
  qp[32 + k] = (bf16)(x1 * __cosf(b) + x0 * __sinf(b));
}

// ---------------- phase 3: q_lat = q_nope @ Wn ----------------

__global__ void __launch_bounds__(128) k_gemm_qlat(const bf16* __restrict__ qnope,
                                                   const bf16* __restrict__ WnT,
                                                   bf16* __restrict__ qlat) {
  int wid  = (blockIdx.x * blockDim.x + threadIdx.x) >> 5;  // 16*128*8
  int lane = threadIdx.x & 31;
  int h = wid / (128 * 8); int rem = wid % (128 * 8);
  int mb = rem / 8, nb = rem % 8;
  int s0 = mb * 16, c0 = nb * 64;
  const bf16* A = qnope + (size_t)h * S_LEN * DN;
  const bf16* B = WnT + (size_t)h * CL * DN;
  v8f acc[4];
#pragma unroll
  for (int i = 0; i < 4; ++i) acc[i] = vzero();
  gemm16x64<DN>(A, DN, s0, B, DN, c0, lane, acc);
  int lo = lane & 15, hi = lane >> 4;
  bf16* out = qlat + (size_t)h * S_LEN * CL;
#pragma unroll
  for (int i = 0; i < 4; ++i)
#pragma unroll
    for (int r = 0; r < 8; ++r)
      out[(size_t)(s0 + r + 8 * hi) * CL + c0 + 16 * i + lo] = (bf16)acc[i][r];
}

// ---------------- phase 4: flash attention over latent dim ----------------
// wave task: (head, 16 query rows, 256-wide c-half). Online softmax, P staged
// through LDS to convert WMMA C-layout -> A-layout for the PV matmuls.
// Score and PV inner loops are software-pipelined one step ahead.

__global__ void __launch_bounds__(128) k_flash(const bf16* __restrict__ qlat,
                                               const bf16* __restrict__ qpe,
                                               const bf16* __restrict__ kvb,
                                               const bf16* __restrict__ kvT,
                                               const bf16* __restrict__ peb,
                                               bf16* __restrict__ xbuf) {
  __shared__ __align__(32) bf16 Pl[4][16][32];
  int wv = threadIdx.x >> 5, lane = threadIdx.x & 31;
  int wid = blockIdx.x * 4 + wv;
  int h = wid >> 8; int rem = wid & 255;
  int sb = rem >> 1; int ch = rem & 1;
  int s0 = sb * 16, cb = ch * 256;
  int lo = lane & 15, hi = lane >> 4;
  const bf16* Aq = qlat + (size_t)h * S_LEN * CL;
  const bf16* Ap = qpe + (size_t)h * S_LEN * DR;

  v8f O[16];
#pragma unroll
  for (int i = 0; i < 16; ++i) O[i] = vzero();
  float m[8], l[8];
#pragma unroll
  for (int r = 0; r < 8; ++r) { m[r] = -3.0e38f; l[r] = 0.0f; }

  int nt = (s0 + 15) / 32 + 1;   // causal: need t <= s0+15
  for (int tt = 0; tt < nt; ++tt) {
    int t0 = tt * 32;
    if (tt + 1 < nt) {  // pull next K/V tiles toward L1 while we compute
      __builtin_prefetch(kvb + (size_t)(t0 + 32 + lo) * CL, 0, 1);
      __builtin_prefetch(kvT + (size_t)(cb + lo * 16) * S_LEN + t0 + 32, 0, 1);
    }

    v8f Sc[2];
    Sc[0] = vzero(); Sc[1] = vzero();
    // ---- scores: pipelined over K=CL, then the small K=DR rope part ----
    {
      v16bf ac = load_a_frag(Aq, CL, s0, 0, lane);
      v16bf b0 = load_b_frag(kvb, CL, t0, 0, lane);
      v16bf b1 = load_b_frag(kvb, CL, t0 + 16, 0, lane);
#pragma unroll 4
      for (int kc = 32; kc < CL; kc += 32) {
        v16bf an = load_a_frag(Aq, CL, s0, kc, lane);
        v16bf b0n = load_b_frag(kvb, CL, t0, kc, lane);
        v16bf b1n = load_b_frag(kvb, CL, t0 + 16, kc, lane);
        Sc[0] = wmma_bf16(ac, b0, Sc[0]);
        Sc[1] = wmma_bf16(ac, b1, Sc[1]);
        ac = an; b0 = b0n; b1 = b1n;
      }
      // overlap last CL step with first rope step's loads
      v16bf ar = load_a_frag(Ap, DR, s0, 0, lane);
      v16bf r0 = load_b_frag(peb, DR, t0, 0, lane);
      v16bf r1 = load_b_frag(peb, DR, t0 + 16, 0, lane);
      Sc[0] = wmma_bf16(ac, b0, Sc[0]);
      Sc[1] = wmma_bf16(ac, b1, Sc[1]);
      v16bf ar2 = load_a_frag(Ap, DR, s0, 32, lane);
      v16bf r0b = load_b_frag(peb, DR, t0, 32, lane);
      v16bf r1b = load_b_frag(peb, DR, t0 + 16, 32, lane);
      Sc[0] = wmma_bf16(ar, r0, Sc[0]);
      Sc[1] = wmma_bf16(ar, r1, Sc[1]);
      Sc[0] = wmma_bf16(ar2, r0b, Sc[0]);
      Sc[1] = wmma_bf16(ar2, r1b, Sc[1]);
    }

    // scale + causal mask (element (row=r+8*hi, t=t0+16j+lo))
#pragma unroll
    for (int j = 0; j < 2; ++j)
#pragma unroll
      for (int r = 0; r < 8; ++r) {
        float v = Sc[j][r] * SM_SCALE;
        if (t0 + 16 * j + lo > s0 + r + 8 * hi) v += MASK_NEG;
        Sc[j][r] = v;
      }

    // online softmax (row lives across the 16 lanes of each half-wave)
    float scv[8];
#pragma unroll
    for (int r = 0; r < 8; ++r) {
      float mx = rowmax16(fmaxf(Sc[0][r], Sc[1][r]));
      float mn = fmaxf(m[r], mx);
      float sc = __expf(m[r] - mn);
      m[r] = mn;
      float p0 = __expf(Sc[0][r] - mn);
      float p1 = __expf(Sc[1][r] - mn);
      Sc[0][r] = p0; Sc[1][r] = p1;
      l[r] = l[r] * sc + rowsum16(p0 + p1);
      scv[r] = sc;
    }
#pragma unroll
    for (int i = 0; i < 16; ++i)
#pragma unroll
      for (int r = 0; r < 8; ++r) O[i][r] *= scv[r];

    // C-layout -> A-layout through wave-private LDS slice
#pragma unroll
    for (int r = 0; r < 8; ++r) {
      int row = r + 8 * hi;
      Pl[wv][row][lo]      = (bf16)Sc[0][r];
      Pl[wv][row][16 + lo] = (bf16)Sc[1][r];
    }
    Frag pa;  // same-wave LDS ordering; compiler inserts s_wait_dscnt
    pa.h[0] = *reinterpret_cast<const v8bf*>(&Pl[wv][lo][8 * hi]);
    pa.h[1] = *reinterpret_cast<const v8bf*>(&Pl[wv][lo][16 + 8 * hi]);

    // ---- PV: pipelined over the 16 c-subtiles ----
    {
      v16bf bcur = load_b_frag(kvT, S_LEN, cb, t0, lane);
#pragma unroll
      for (int i = 0; i < 15; ++i) {
        v16bf bn = load_b_frag(kvT, S_LEN, cb + 16 * (i + 1), t0, lane);
        O[i] = wmma_bf16(pa.v, bcur, O[i]);
        bcur = bn;
      }
      O[15] = wmma_bf16(pa.v, bcur, O[15]);
    }
  }

  float rinv[8];
#pragma unroll
  for (int r = 0; r < 8; ++r) rinv[r] = __frcp_rn(l[r]);
  bf16* X = xbuf + (size_t)h * S_LEN * CL;
#pragma unroll
  for (int i = 0; i < 16; ++i)
#pragma unroll
    for (int r = 0; r < 8; ++r) {
      int row = r + 8 * hi;
      X[(size_t)(s0 + row) * CL + cb + 16 * i + lo] = (bf16)(O[i][r] * rinv[r]);
    }
}

// ---------------- phase 5: out = x @ Wv^T ----------------

__global__ void __launch_bounds__(128) k_gemm_out(const bf16* __restrict__ xbuf,
                                                  const bf16* __restrict__ Wv,
                                                  float* __restrict__ out) {
  int wid  = (blockIdx.x * blockDim.x + threadIdx.x) >> 5;  // 16*128*2
  int lane = threadIdx.x & 31;
  int h = wid / (128 * 2); int rem = wid % (128 * 2);
  int mb = rem / 2, nb = rem % 2;
  int s0 = mb * 16, d0 = nb * 64;
  const bf16* A = xbuf + (size_t)h * S_LEN * CL;
  const bf16* B = Wv + (size_t)h * DV * CL;
  v8f acc[4];
#pragma unroll
  for (int i = 0; i < 4; ++i) acc[i] = vzero();
  gemm16x64<CL>(A, CL, s0, B, CL, d0, lane, acc);
  int lo = lane & 15, hi = lane >> 4;
#pragma unroll
  for (int i = 0; i < 4; ++i)
#pragma unroll
    for (int r = 0; r < 8; ++r)
      out[((size_t)(s0 + r + 8 * hi) * HEADS + h) * DV + d0 + 16 * i + lo] = acc[i][r];
}

// ---------------- host launcher ----------------

extern "C" void kernel_launch(void* const* d_in, const int* in_sizes, int n_in,
                              void* d_out, int out_size, void* d_ws, size_t ws_size,
                              hipStream_t stream) {
  (void)in_sizes; (void)n_in; (void)out_size; (void)ws_size;
  const float* WUQ       = (const float*)d_in[0];
  const float* WUQ_scale = (const float*)d_in[1];
  const float* WUKV      = (const float*)d_in[2];
  const float* WUKV_scale= (const float*)d_in[3];
  const float* Q         = (const float*)d_in[4];
  const float* KV        = (const float*)d_in[5];
  const float* PE        = (const float*)d_in[6];
  const float* idx_theta = (const float*)d_in[9];
  float* out = (float*)d_out;

  char* base = (char*)d_ws;
  size_t o = 0;
  auto arena = [&](size_t bytes) { char* p = base + o; o = (o + bytes + 255) & ~(size_t)255; return p; };
  bf16* WUQd  = (bf16*)arena((size_t)QROWS * QL * 2);
  bf16* WnT   = (bf16*)arena((size_t)HEADS * CL * DN * 2);
  bf16* Wv    = (bf16*)arena((size_t)HEADS * DV * CL * 2);
  bf16* Qb    = (bf16*)arena((size_t)S_LEN * QL * 2);
  bf16* kvb   = (bf16*)arena((size_t)S_LEN * CL * 2);
  bf16* kvT   = (bf16*)arena((size_t)CL * S_LEN * 2);
  bf16* peb   = (bf16*)arena((size_t)S_LEN * DR * 2);
  float* qfull= (float*)arena((size_t)S_LEN * QROWS * 4);
  bf16* qnope = (bf16*)arena((size_t)HEADS * S_LEN * DN * 2);
  bf16* qpe   = (bf16*)arena((size_t)HEADS * S_LEN * DR * 2);
  bf16* qlat  = (bf16*)arena((size_t)HEADS * S_LEN * CL * 2);
  bf16* xbuf  = (bf16*)arena((size_t)HEADS * S_LEN * CL * 2);

  k_dequant_wuq <<<(QROWS * QL) / 256, 256, 0, stream>>>(WUQ, WUQ_scale, WUQd);
  k_dequant_wukv<<<(HEADS * 256 * CL) / 256, 256, 0, stream>>>(WUKV, WUKV_scale, WnT, Wv);
  k_conv_q      <<<(S_LEN * QL) / 256, 256, 0, stream>>>(Q, Qb);
  k_conv_kv     <<<(S_LEN * CL) / 256, 256, 0, stream>>>(KV, kvb, kvT);
  k_rope_pe     <<<(S_LEN * 32) / 256, 256, 0, stream>>>(PE, idx_theta, peb);
  k_gemm_q      <<<(128 * 48) / 4, 128, 0, stream>>>(Qb, WUQd, qfull);
  k_rope_split  <<<(HEADS * S_LEN * 32) / 256, 256, 0, stream>>>(qfull, idx_theta, qnope, qpe);
  k_gemm_qlat   <<<(HEADS * 128 * 8) / 4, 128, 0, stream>>>(qnope, WnT, qlat);
  k_flash       <<<(HEADS * 128 * 2) / 4, 128, 0, stream>>>(qlat, qpe, kvb, kvT, peb, xbuf);
  k_gemm_out    <<<(HEADS * 128 * 2) / 4, 128, 0, stream>>>(xbuf, Wv, out);
}
